// LstmTrainStep_70660801954473
// MI455X (gfx1250) — matrix operand order repaired and checked by reading, not compiled, not run on hardware
//
#include <hip/hip_runtime.h>
#include <hip/hip_bf16.h>
#include <stdint.h>

typedef __bf16 bf16_t;
typedef __attribute__((ext_vector_type(16))) __bf16 v16bf;
typedef __attribute__((ext_vector_type(8)))  float  v8f;

#define T_STEPS 512
#define BATCH   128
#define INDIM   512
#define HID     512
#define GATES   2048   /* 4*HID */
#define KTOT    1024   /* INDIM + HID */

/* workspace layout (bytes) */
#define WS_W     0                              /* 2048*1024 bf16 = 4 MiB  */
#define WS_BIAS  (WS_W + GATES*KTOT*2)          /* 2048 f32                */
#define WS_C     (WS_BIAS + GATES*4)            /* B*H f32 = 256 KiB       */
#define WS_H0    (WS_C + BATCH*HID*4)           /* packed h buf0, 128 KiB  */
#define WS_H1    (WS_H0 + BATCH*HID*2)          /* packed h buf1, 128 KiB  */
#define WS_X     (WS_H1 + BATCH*HID*2)          /* packed latent, 64 MiB   */

/* ---- fragment slot maps (16-bit WMMA layouts, wave32) ----
 * A (16x32, M x K):  lane = M + (K&8 ? 16 : 0),  e = (K&7) + (K&16 ? 8 : 0)
 * B (32x16, K x N):  lane = N + (K&8 ? 16 : 0),  e = (K&7) + (K&16 ? 8 : 0)
 * slot = lane*16 + e  (each lane's 16 bf16 are contiguous -> 32B loads)
 */

__global__ void prep_weights(const float* __restrict__ w_ih,
                             const float* __restrict__ w_hh,
                             const float* __restrict__ b_ih,
                             const float* __restrict__ b_hh,
                             bf16_t* __restrict__ Wp,
                             float* __restrict__ bias) {
    int idx = blockIdx.x * 256 + threadIdx.x;          /* 2,097,152 slots */
    int e    = idx & 15;
    int lane = (idx >> 4) & 31;
    int kt   = (idx >> 9) & 31;                        /* k tile, 32 per column tile */
    int ct   = idx >> 14;                              /* column tile, 0..127 */
    int n    = lane & 15;
    int K    = ((e < 8) ? e : e + 8) + ((lane >= 16) ? 8 : 0);
    int gc   = ct * 16 + n;                            /* gate column 0..2047 */
    int kg   = kt * 32 + K;                            /* global K 0..1023 */
    float v  = (kg < INDIM) ? w_ih[gc * INDIM + kg]
                            : w_hh[gc * HID + (kg - INDIM)];
    Wp[idx] = (bf16_t)v;
    if (idx < GATES) bias[idx] = b_ih[idx] + b_hh[idx];
}

__global__ void prep_latent(const float* __restrict__ latent,
                            bf16_t* __restrict__ Xp) {
    int idx  = blockIdx.x * 256 + threadIdx.x;         /* 33,554,432 slots */
    int e    = idx & 15;
    int lane = (idx >> 4) & 31;
    int kt   = (idx >> 9) & 15;                        /* k tile within x (0..15) */
    int rt   = (idx >> 13) & 7;                        /* row tile (0..7) */
    int t    = idx >> 16;
    int M    = lane & 15;
    int K    = ((e < 8) ? e : e + 8) + ((lane >= 16) ? 8 : 0);
    int b    = rt * 16 + M;
    int k    = kt * 32 + K;
    Xp[idx]  = (bf16_t)latent[((size_t)t * BATCH + b) * INDIM + k];
}

/* apply step-0 mask/noise to the initial state; pack h into A-fragment buf0 */
__global__ void prep_state(const float* __restrict__ h0,
                           const float* __restrict__ c0,
                           const uint8_t* __restrict__ rm,
                           const uint8_t* __restrict__ cm,
                           const float* __restrict__ stds,
                           const float* __restrict__ noise,
                           float* __restrict__ c_state,
                           bf16_t* __restrict__ hpack0) {
    int idx = blockIdx.x * 256 + threadIdx.x;          /* 65536 */
    int b = idx / HID, j = idx % HID;
    bool m   = rm[b * T_STEPS] || cm[b * T_STEPS];
    float nz = noise[b * HID + j] * stds[b * T_STEPS];
    float h  = (m ? 0.f : h0[idx]) + nz;
    float c  = (m ? 0.f : c0[idx]) + nz;
    c_state[idx] = c;
    int rt = b >> 4, M = b & 15, kt = j >> 5, K = j & 31;
    int lane = M + ((K & 8) ? 16 : 0);
    int e    = (K & 7) + ((K & 16) ? 8 : 0);
    hpack0[(rt * 16 + kt) * 512 + lane * 16 + e] = (bf16_t)h;
}

__device__ __forceinline__ float sigm(float x) { return 1.f / (1.f + __expf(-x)); }

/* one recurrence step: gates = [x_t,h] @ W^T (+bias), cell update, fused
 * mask/noise for step t+1, repack h into the other double buffer. */
__global__ __launch_bounds__(128) void lstm_step(
        const bf16_t* __restrict__ Wp, const float* __restrict__ bias,
        const bf16_t* __restrict__ Xp,
        const bf16_t* __restrict__ h_in, bf16_t* __restrict__ h_out,
        float* __restrict__ c_state,
        const uint8_t* __restrict__ rm, const uint8_t* __restrict__ cm,
        const float* __restrict__ stds, const float* __restrict__ noise,
        float* __restrict__ out, int t) {
    const int lane = threadIdx.x & 31;
    const int gate = threadIdx.x >> 5;    /* 0..3 -> i,f,g,o */
    const int jt   = blockIdx.x;          /* 0..31 : H column tile */
    const int rt   = blockIdx.y;          /* 0..7  : batch row tile */

    v8f acc = {};
    const int ct = gate * 32 + jt;        /* weight column tile */
    const bf16_t* wp = Wp + ((size_t)ct * 32) * 512 + lane * 16;
    const bf16_t* xp = Xp + ((size_t)(t * 8 + rt) * 16) * 512 + lane * 16;
    const bf16_t* hp = h_in + ((size_t)rt * 16) * 512 + lane * 16;

#pragma unroll
    for (int kt = 0; kt < 16; ++kt) {      /* K = 0..511 : x part */
        v16bf a = *(const v16bf*)(xp + (size_t)kt * 512);
        v16bf b = *(const v16bf*)(wp + (size_t)kt * 512);
        acc = __builtin_amdgcn_wmma_f32_16x16x32_bf16(false, a, false, b,
                                                      (short)0, acc, false, false);
    }
#pragma unroll
    for (int kt = 0; kt < 16; ++kt) {      /* K = 512..1023 : h part */
        v16bf a = *(const v16bf*)(hp + (size_t)kt * 512);
        v16bf b = *(const v16bf*)(wp + (size_t)(16 + kt) * 512);
        acc = __builtin_amdgcn_wmma_f32_16x16x32_bf16(false, a, false, b,
                                                      (short)0, acc, false, false);
    }

    /* exchange the 4 gate tiles through LDS (C layout: VGPR v, lane l ->
       M = v + 8*(l>=16), N = l&15) */
    __shared__ float g_lds[4][16][17];
    {
        int M0 = (lane >= 16) ? 8 : 0;
        int N  = lane & 15;
        float bv = bias[gate * HID + jt * 16 + N];
#pragma unroll
        for (int v = 0; v < 8; ++v) g_lds[gate][M0 + v][N] = acc[v] + bv;
    }
    __syncthreads();

    const int tid = threadIdx.x;
#pragma unroll
    for (int q = 0; q < 2; ++q) {
        int e = tid * 2 + q;               /* 0..255 element of the 16x16 tile */
        int M = e >> 4, N = e & 15;
        float gi = g_lds[0][M][N], gf = g_lds[1][M][N];
        float gg = g_lds[2][M][N], go = g_lds[3][M][N];
        int row = rt * 16 + M;
        int col = jt * 16 + N;
        float c = sigm(gf) * c_state[row * HID + col] + sigm(gi) * tanhf(gg);
        float h = sigm(go) * tanhf(c);
        out[(size_t)t * BATCH * HID + row * HID + col] = h;   /* hidden[t] (pre-mask) */

        if (t + 1 < T_STEPS) {
            bool m   = rm[row * T_STEPS + t + 1] || cm[row * T_STEPS + t + 1];
            float nz = noise[(size_t)(t + 1) * BATCH * HID + row * HID + col]
                       * stds[row * T_STEPS + t + 1];
            float hn = (m ? 0.f : h) + nz;
            float cn = (m ? 0.f : c) + nz;
            c_state[row * HID + col] = cn;
            int kt2 = col >> 5, K = col & 31;
            int lw  = M + ((K & 8) ? 16 : 0);
            int ee  = (K & 7) + ((K & 16) ? 8 : 0);
            h_out[(rt * 16 + kt2) * 512 + lw * 16 + ee] = (bf16_t)hn;
        } else {
            size_t base = (size_t)T_STEPS * BATCH * HID;
            out[base + row * HID + col] = h;                   /* hN */
            out[base + BATCH * HID + row * HID + col] = c;     /* cN */
        }
    }
}

extern "C" void kernel_launch(void* const* d_in, const int* in_sizes, int n_in,
                              void* d_out, int out_size, void* d_ws, size_t ws_size,
                              hipStream_t stream) {
    const float*   latent = (const float*)d_in[0];
    const float*   h_t    = (const float*)d_in[1];
    const float*   c_t    = (const float*)d_in[2];
    const uint8_t* reset  = (const uint8_t*)d_in[3];   /* bool [1,B,T,1] */
    const uint8_t* clearm = (const uint8_t*)d_in[4];   /* bool [B,T]     */
    const float*   stds   = (const float*)d_in[5];
    const float*   noise  = (const float*)d_in[6];
    const float*   w_ih   = (const float*)d_in[7];
    const float*   w_hh   = (const float*)d_in[8];
    const float*   b_ih   = (const float*)d_in[9];
    const float*   b_hh   = (const float*)d_in[10];
    (void)in_sizes; (void)n_in; (void)out_size; (void)ws_size;

    char* ws = (char*)d_ws;
    bf16_t* Wp      = (bf16_t*)(ws + WS_W);
    float*  bias    = (float*) (ws + WS_BIAS);
    float*  c_state = (float*) (ws + WS_C);
    bf16_t* h0p     = (bf16_t*)(ws + WS_H0);
    bf16_t* h1p     = (bf16_t*)(ws + WS_H1);
    bf16_t* Xp      = (bf16_t*)(ws + WS_X);
    float*  out     = (float*)d_out;

    prep_weights<<<(GATES * KTOT) / 256, 256, 0, stream>>>(w_ih, w_hh, b_ih, b_hh, Wp, bias);
    prep_latent<<<(T_STEPS * BATCH * INDIM) / 256, 256, 0, stream>>>(latent, Xp);
    prep_state<<<(BATCH * HID) / 256, 256, 0, stream>>>(h_t, c_t, reset, clearm,
                                                        stds, noise, c_state, h0p);
    for (int t = 0; t < T_STEPS; ++t) {
        bf16_t* hin  = (t & 1) ? h1p : h0p;
        bf16_t* hout = (t & 1) ? h0p : h1p;
        lstm_step<<<dim3(32, 8), 128, 0, stream>>>(Wp, bias, Xp, hin, hout, c_state,
                                                   reset, clearm, stds, noise, out, t);
    }
}
